// PyTorchMIMN_21320217657598
// MI455X (gfx1250) — matrix operand order repaired and checked by reading, not compiled, not run on hardware
//
#include <hip/hip_runtime.h>
#include <hip/hip_bf16.h>

// ---------------------------------------------------------------------------
// MIMN on MI455X (gfx1250, wave32, WMMA bf16 16x16x32, LDS-resident weights)
// Round 2: hoisted A-fragments (register-resident across N-tiles), activation
// fused into GEMM2 store (one fewer LDS pass + barrier per recurrence step).
// ---------------------------------------------------------------------------
#define Vn   100000
#define En   64
#define Sn   4
#define Bn   2048
#define Tn   200
#define Dn   128      // 2*E
#define Cn   64
#define CINn 256      // 2*E + D
#define Pn   514
#define PPAD 528      // Pn padded to multiple of 16
#define G3   192      // 3*C

// workspace layout (bytes)
#define OFF_WIH 0                       // bf16 [192][256]   98304
#define OFF_WHH 98304                   // bf16 [192][64]    24576
#define OFF_WP  122880                  // bf16 [528][64]    67584
#define OFF_WO  190464                  // bf16 [64][192]    24576
#define OFF_FC1 215040                  // bf16 [208][320]  133120
#define OFF_FC2 348160                  // bf16 [80][224]    35840
#define OFF_HS  384000                  // f32  [2048][128] 1048576
#define OFF_LO  1432576                 // f32  [2048][64]   524288

typedef __attribute__((ext_vector_type(16))) __bf16 v16bf;
typedef __attribute__((ext_vector_type(8)))  float  v8f;
typedef __attribute__((ext_vector_type(4)))  unsigned uv4;

union V16 { v16bf v; uv4 q[2]; };

__device__ __forceinline__ unsigned short f2bf(float f) {
  union { float f; unsigned u; } v; v.f = f;
  const unsigned r = 0x7FFFu + ((v.u >> 16) & 1u);
  return (unsigned short)((v.u + r) >> 16);
}
__device__ __forceinline__ float sigm(float x) { return 1.f / (1.f + expf(-x)); }
__device__ __forceinline__ float softplusf_(float x) { return (x > 20.f) ? x : log1pf(expf(x)); }
__device__ __forceinline__ float red8(float x) {
  x += __shfl_xor(x, 1, 8);
  x += __shfl_xor(x, 2, 8);
  x += __shfl_xor(x, 4, 8);
  return x;
}

// Load A fragments (16xK tile, bf16 in LDS, row stride lda halfwords) into
// registers once; ISA 16-bit A 16x32 layout: lane l<16 holds K[k0..k0+7] and
// K[k0+16..k0+23]; lanes 16-31 hold K[k0+8..15] and K[k0+24..31] of row l&15.
__device__ __forceinline__ void load_afrag(const unsigned short* A, int lda,
                                           int ktiles, V16* out) {
  const int lane = threadIdx.x & 31;
  const int l16  = lane & 15;
  const int half = lane >> 4;
  const unsigned short* ap = A + l16 * lda + half * 8;
  for (int k = 0; k < ktiles; ++k) {
    out[k].q[0] = *(const uv4*)(ap);
    out[k].q[1] = *(const uv4*)(ap + 16);
    ap += 32;
  }
}

// C += A * W^T with pre-loaded A fragments; W is [N][K] bf16 row-major in LDS.
// B operand: lane n<16 holds column n, K[k0..k0+7] and K[k0+16..23]; lanes 16-31 K[k0+8..15] and K[k0+24..31].
__device__ __forceinline__ v8f wmma_b(const V16* a, const unsigned short* Brow,
                                      int ldb, int ktiles, v8f acc) {
  const int lane = threadIdx.x & 31;
  const int l16  = lane & 15;
  const int half = lane >> 4;
  const unsigned short* bp = Brow + l16 * ldb + half * 8;
  for (int k = 0; k < ktiles; ++k) {
    V16 b;
    b.q[0] = *(const uv4*)(bp);
    b.q[1] = *(const uv4*)(bp + 16);
    acc = __builtin_amdgcn_wmma_f32_16x16x32_bf16(false, a[k].v, false, b.v,
                                                  (short)0, acc, false, false);
    bp += 32;
  }
  return acc;
}

// ---------------------------------------------------------------------------
// fp32 -> bf16 weight prep with zero padding
// ---------------------------------------------------------------------------
__global__ void prep_bf16(const float* __restrict__ src, unsigned short* __restrict__ dst,
                          int rows, int cols, int drows, int dcols) {
  int i = blockIdx.x * blockDim.x + threadIdx.x;
  if (i >= drows * dcols) return;
  int r = i / dcols, c = i % dcols;
  float v = (r < rows && c < cols) ? src[r * cols + c] : 0.f;
  dst[i] = f2bf(v);
}

// ---------------------------------------------------------------------------
// hist_sum: masked sum of hist embeddings, (B,128) fp32
// ---------------------------------------------------------------------------
__global__ __launch_bounds__(128) void hist_sum_kernel(
    const int* __restrict__ hi, const int* __restrict__ hc,
    const float* __restrict__ mask, const float* __restrict__ emb,
    float* __restrict__ hs) {
  const int b = blockIdx.x, c = threadIdx.x;
  const int* hir = hi + b * Tn;
  const int* hcr = hc + b * Tn;
  const float* mr = mask + b * Tn;
  const int cc = c & 63;
  float acc = 0.f;
  for (int t = 0; t < Tn; ++t) {
    const int idx = (c < 64) ? hir[t] : hcr[t];
    acc += mr[t] * emb[idx * 64 + cc];
  }
  hs[b * 128 + c] = acc;
}

// ---------------------------------------------------------------------------
// RNN: one workgroup = 16 batch rows, 4 waves, full T=200 recurrence.
// Weights live in LDS (bf16). last_out = [h_final, read_final] @ Wo^T + bo.
// ---------------------------------------------------------------------------
__global__ __launch_bounds__(128) void mimn_rnn_kernel(
    const int* __restrict__ hitem, const int* __restrict__ hcate,
    const float* __restrict__ mask, const float* __restrict__ emb,
    const unsigned short* __restrict__ wpack,   // Wih|Whh|Wp bf16 contiguous
    const unsigned short* __restrict__ wWo,     // Wo bf16 [64][192]
    const float* __restrict__ bih, const float* __restrict__ bhh,
    const float* __restrict__ bp,  const float* __restrict__ bo,
    const float* __restrict__ M0,  float* __restrict__ lastOut) {
  extern __shared__ char smem[];
  unsigned short* sWih  = (unsigned short*)(smem);              // [192][256]
  unsigned short* sWhh  = (unsigned short*)(smem + 98304);      // [192][64]
  unsigned short* sWp   = (unsigned short*)(smem + 122880);     // [528][64]
  unsigned short* sX    = (unsigned short*)(smem + 190464);     // [16][256] bf16
  unsigned short* sHbf  = (unsigned short*)(smem + 198656);     // [16][64] committed h
  unsigned short* sHnbf = (unsigned short*)(smem + 200704);     // [16][64] h_new
  float* sH    = (float*)(smem + 202752);                       // [16][64]
  float* sRead = (float*)(smem + 206848);                       // [16][128]
  float* sM    = (float*)(smem + 215040);                       // [16][4][128]
  float* sGi   = (float*)(smem + 247808);                       // [16][192]
  float* sGh   = (float*)(smem + 260096);                       // [16][192]
  float* sP    = (float*)(smem + 272384);                       // [16][528]
  int*   sI1   = (int*)(smem + 306176);
  int*   sI2   = sI1 + 16;
  float* sMk   = (float*)(sI2 + 16);

  const int tid  = threadIdx.x;
  const int wave = tid >> 5;
  const int lane = tid & 31;
  const int l16  = lane & 15;
  const int rb   = (lane >> 4) * 8;
  const int b0   = blockIdx.x * 16;

  // Stage Wih|Whh|Wp into LDS (190464 B contiguous)
  {
    uv4* dst = (uv4*)smem;
    const uv4* src = (const uv4*)wpack;
    for (int i = tid; i < 190464 / 16; i += 128) dst[i] = src[i];
  }
  for (int i = tid; i < 16 * 64; i += 128) { sH[i] = 0.f; sHbf[i] = 0; sHnbf[i] = 0; }
  for (int i = tid; i < 16 * 128; i += 128) sRead[i] = 0.f;
  for (int i = tid; i < 16 * 512; i += 128) sM[i] = M0[i & 511];
  __syncthreads();

  for (int t = 0; t < Tn; ++t) {
    if (tid < 16) {
      const int b = b0 + tid;
      sI1[tid] = hitem[b * Tn + t];
      sI2[tid] = hcate[b * Tn + t];
      sMk[tid] = mask[b * Tn + t];
      if (t + 1 < Tn) {  // gfx1250 global_prefetch of next step's embedding rows
        __builtin_prefetch(emb + hitem[b * Tn + t + 1] * 64, 0, 1);
        __builtin_prefetch(emb + hcate[b * Tn + t + 1] * 64, 0, 1);
      }
    }
    __syncthreads();
    // ctrl_in = [emb_item | emb_cate | read] -> bf16 tile
    for (int i = tid; i < 16 * 256; i += 128) {
      const int r = i >> 8, c = i & 255;
      float v;
      if (c < 64)       v = emb[sI1[r] * 64 + c];
      else if (c < 128) v = emb[sI2[r] * 64 + (c - 64)];
      else              v = sRead[r * 128 + (c - 128)];
      sX[i] = f2bf(v);
    }
    __syncthreads();
    // GEMM1: gi = X @ Wih^T + bih ; gh = h @ Whh^T + bhh   (N=192, 12 tiles)
    {
      V16 aX[8], aH[2];                 // A fragments hoisted: loaded once,
      load_afrag(sX,   256, 8, aX);     // reused across all 3 N-tiles/wave
      load_afrag(sHbf,  64, 2, aH);
      for (int nt = wave; nt < 12; nt += 4) {
        v8f gi = {}; v8f gh = {};
        gi = wmma_b(aX, sWih + nt * 16 * 256, 256, 8, gi);
        gh = wmma_b(aH, sWhh + nt * 16 * 64,   64, 2, gh);
        const int col = nt * 16 + l16;
        const float bi = bih[col], bh = bhh[col];
#pragma unroll
        for (int r = 0; r < 8; ++r) {
          sGi[(rb + r) * 192 + col] = gi[r] + bi;
          sGh[(rb + r) * 192 + col] = gh[r] + bh;
        }
      }
    }
    __syncthreads();
    // GRU elementwise
    for (int i = tid; i < 16 * 64; i += 128) {
      const int r = i >> 6, c = i & 63;
      const float rg = sigm(sGi[r * 192 + c] + sGh[r * 192 + c]);
      const float zg = sigm(sGi[r * 192 + 64 + c] + sGh[r * 192 + 64 + c]);
      const float ng = tanhf(sGi[r * 192 + 128 + c] + rg * sGh[r * 192 + 128 + c]);
      const float hold = sH[i];
      const float hn = (1.f - zg) * ng + zg * hold;
      const float hc = (sMk[r] > 0.f) ? hn : hold;
      sH[i]    = hc;
      sHbf[i]  = f2bf(hc);
      sHnbf[i] = f2bf(hn);
    }
    __syncthreads();
    // GEMM2: p = h_new @ Wp^T + bp, activation fused into the store
    {
      V16 aN[2];
      load_afrag(sHnbf, 64, 2, aN);
      for (int nt = wave; nt < 33; nt += 4) {
        v8f p = {};
        p = wmma_b(aN, sWp + nt * 16 * 64, 64, 2, p);
        const int col = nt * 16 + l16;
        const float bb = (col < Pn) ? bp[col] : 0.f;
#pragma unroll
        for (int r = 0; r < 8; ++r) {
          const float v = p[r] + bb;
          float o;
          if (col == 128 || col == 257) o = softplusf_(v);   // beta_r, beta_w
          else if (col < 257)           o = tanhf(v);        // k_r, k_w
          else if (col < 386)           o = sigm(v);         // erase
          else                          o = tanhf(v);        // add
          sP[(rb + r) * PPAD + col] = o;
        }
      }
    }
    __syncthreads();
    // memory network: 8 threads per row, shfl-xor reductions over D=128
    {
      const int r = tid >> 3, g = tid & 7;
      float* pr = sP + r * PPAD;
      float* Mr = sM + r * 512;
      float k2 = 0.f, w2 = 0.f;
      float dK[4] = {0, 0, 0, 0}, dW[4] = {0, 0, 0, 0}, nM[4] = {0, 0, 0, 0};
      for (int j = 0; j < 16; ++j) {
        const int d = g + 8 * j;
        const float kr = pr[d], kw = pr[129 + d];
        k2 += kr * kr; w2 += kw * kw;
#pragma unroll
        for (int s = 0; s < 4; ++s) {
          const float m = Mr[s * 128 + d];
          dK[s] += kr * m; dW[s] += kw * m; nM[s] += m * m;
        }
      }
      k2 = red8(k2); w2 = red8(w2);
#pragma unroll
      for (int s = 0; s < 4; ++s) { dK[s] = red8(dK[s]); dW[s] = red8(dW[s]); nM[s] = red8(nM[s]); }
      const float nk = sqrtf(k2) + 1e-8f, nw = sqrtf(w2) + 1e-8f;
      const float betaR = pr[128], betaW = pr[257];
      float lr[4], lw[4];
#pragma unroll
      for (int s = 0; s < 4; ++s) {
        const float nm = sqrtf(nM[s]) + 1e-8f;
        lr[s] = betaR * dK[s] / (nk * nm);
        lw[s] = betaW * dW[s] / (nw * nm);
      }
      const float mr = fmaxf(fmaxf(lr[0], lr[1]), fmaxf(lr[2], lr[3]));
      const float mw = fmaxf(fmaxf(lw[0], lw[1]), fmaxf(lw[2], lw[3]));
      float sr = 0.f, sw = 0.f;
#pragma unroll
      for (int s = 0; s < 4; ++s) {
        lr[s] = expf(lr[s] - mr); lw[s] = expf(lw[s] - mw);
        sr += lr[s]; sw += lw[s];
      }
      const float inr = 1.f / sr, inw = 1.f / sw;
      const bool valid = sMk[r] > 0.f;
      for (int j = 0; j < 16; ++j) {
        const int d = g + 8 * j;
        float rd = 0.f;
#pragma unroll
        for (int s = 0; s < 4; ++s) rd += (lr[s] * inr) * Mr[s * 128 + d];
        const float er = pr[258 + d], ad = pr[386 + d];
        if (valid) {
          sRead[r * 128 + d] = rd;
#pragma unroll
          for (int s = 0; s < 4; ++s) {
            const float w = lw[s] * inw;
            Mr[s * 128 + d] = Mr[s * 128 + d] * (1.f - w * er) + w * ad;
          }
        }
      }
    }
    __syncthreads();
  }

  // last_out = [h_final | read_final] @ Wo^T + bo  (one GEMM instead of 200)
  for (int i = tid; i < 16 * 192; i += 128) {
    const int r = i / 192, c = i % 192;
    const float v = (c < 64) ? sH[r * 64 + c] : sRead[r * 128 + (c - 64)];
    sX[i] = f2bf(v);                       // reuse sX as [16][192] bf16
  }
  {
    uv4* dst = (uv4*)sP;                   // reuse sP to stage Wo bf16 [64][192]
    const uv4* src = (const uv4*)wWo;
    for (int i = tid; i < 24576 / 16; i += 128) dst[i] = src[i];
  }
  __syncthreads();
  {
    const unsigned short* sWo = (const unsigned short*)sP;
    V16 aO[6];
    load_afrag(sX, 192, 6, aO);
    v8f o = {};
    o = wmma_b(aO, sWo + wave * 16 * 192, 192, 6, o);
    const int col = wave * 16 + l16;
    const float bb = bo[col];
#pragma unroll
    for (int r = 0; r < 8; ++r) lastOut[(b0 + rb + r) * 64 + col] = o[r] + bb;
  }
}

// ---------------------------------------------------------------------------
// MLP head: x = [item_emb | hist_sum | last_out] (320) -> fc1(200) prelu
//           -> fc2(80) prelu -> fc3(2)
// ---------------------------------------------------------------------------
__global__ __launch_bounds__(128) void mimn_mlp_kernel(
    const int* __restrict__ item, const int* __restrict__ cate,
    const float* __restrict__ emb,
    const float* __restrict__ hs, const float* __restrict__ lastOut,
    const unsigned short* __restrict__ fcpack,   // fc1|fc2 bf16 contiguous
    const float* __restrict__ fc1b, const float* __restrict__ p1a,
    const float* __restrict__ fc2b, const float* __restrict__ p2a,
    const float* __restrict__ fc3w, const float* __restrict__ fc3b,
    float* __restrict__ out) {
  extern __shared__ char smem[];
  unsigned short* sF1 = (unsigned short*)smem;              // [208][320]
  unsigned short* sF2 = (unsigned short*)(smem + 133120);   // [80][224]
  unsigned short* sXb = (unsigned short*)(smem + 168960);   // [16][320]
  unsigned short* sA1 = (unsigned short*)(smem + 179200);   // [16][224] bf16
  float*          sA2 = (float*)(smem + 186368);            // [16][80]

  const int tid  = threadIdx.x;
  const int wave = tid >> 5;
  const int lane = tid & 31;
  const int l16  = lane & 15;
  const int rb   = (lane >> 4) * 8;
  const int b0   = blockIdx.x * 16;

  {
    uv4* dst = (uv4*)smem;
    const uv4* src = (const uv4*)fcpack;
    for (int i = tid; i < 168960 / 16; i += 128) dst[i] = src[i];
  }
  for (int i = tid; i < 16 * 320; i += 128) {
    const int r = i / 320, c = i % 320;
    const int b = b0 + r;
    float v;
    if (c < 64)        v = emb[item[b] * 64 + c];
    else if (c < 128)  v = emb[cate[b] * 64 + (c - 64)];
    else if (c < 256)  v = hs[b * 128 + (c - 128)];
    else               v = lastOut[b * 64 + (c - 256)];
    sXb[i] = f2bf(v);
  }
  for (int i = tid; i < 16 * 224; i += 128) sA1[i] = 0;
  __syncthreads();

  const float a1 = p1a[0];
  {
    V16 aX[10];
    load_afrag(sXb, 320, 10, aX);
    for (int nt = wave; nt < 13; nt += 4) {     // N=208 padded (200 real)
      v8f a = {};
      a = wmma_b(aX, sF1 + nt * 16 * 320, 320, 10, a);
      const int col = nt * 16 + l16;
      const float bb = (col < 200) ? fc1b[col] : 0.f;
#pragma unroll
      for (int r = 0; r < 8; ++r) {
        float v = a[r] + bb;
        v = (v > 0.f) ? v : a1 * v;
        sA1[(rb + r) * 224 + col] = (col < 200) ? f2bf(v) : (unsigned short)0;
      }
    }
  }
  __syncthreads();

  const float a2 = p2a[0];
  {
    V16 aA[7];
    load_afrag(sA1, 224, 7, aA);
    for (int nt = wave; nt < 5; nt += 4) {      // N=80, K=224 (padded, zeros)
      v8f a = {};
      a = wmma_b(aA, sF2 + nt * 16 * 224, 224, 7, a);
      const int col = nt * 16 + l16;
      const float bb = fc2b[col];
#pragma unroll
      for (int r = 0; r < 8; ++r) {
        float v = a[r] + bb;
        v = (v > 0.f) ? v : a2 * v;
        sA2[(rb + r) * 80 + col] = v;
      }
    }
  }
  __syncthreads();

  if (tid < 32) {                               // fc3: 2 outputs of 80
    const int r = tid >> 1, o = tid & 1;
    float acc = fc3b[o];
    for (int k = 0; k < 80; ++k) acc += sA2[r * 80 + k] * fc3w[o * 80 + k];
    out[(b0 + r) * 2 + o] = acc;
  }
}

// ---------------------------------------------------------------------------
extern "C" void kernel_launch(void* const* d_in, const int* in_sizes, int n_in,
                              void* d_out, int out_size, void* d_ws, size_t ws_size,
                              hipStream_t stream) {
  (void)in_sizes; (void)n_in; (void)out_size; (void)ws_size;
  const int*   item  = (const int*)d_in[0];
  const int*   cate  = (const int*)d_in[1];
  const int*   hitem = (const int*)d_in[2];
  const int*   hcate = (const int*)d_in[3];
  const float* mask  = (const float*)d_in[4];
  const float* emb   = (const float*)d_in[5];
  const float* Wih   = (const float*)d_in[6];
  const float* Whh   = (const float*)d_in[7];
  const float* bih   = (const float*)d_in[8];
  const float* bhh   = (const float*)d_in[9];
  const float* Wp    = (const float*)d_in[10];
  const float* bp    = (const float*)d_in[11];
  const float* Wo    = (const float*)d_in[12];
  const float* bo    = (const float*)d_in[13];
  const float* M0    = (const float*)d_in[14];
  const float* fc1w  = (const float*)d_in[15];
  const float* fc1b  = (const float*)d_in[16];
  const float* p1a   = (const float*)d_in[17];
  const float* fc2w  = (const float*)d_in[18];
  const float* fc2b  = (const float*)d_in[19];
  const float* p2a   = (const float*)d_in[20];
  const float* fc3w  = (const float*)d_in[21];
  const float* fc3b  = (const float*)d_in[22];

  char* ws = (char*)d_ws;
  unsigned short* wWih = (unsigned short*)(ws + OFF_WIH);
  unsigned short* wWhh = (unsigned short*)(ws + OFF_WHH);
  unsigned short* wWp  = (unsigned short*)(ws + OFF_WP);
  unsigned short* wWo  = (unsigned short*)(ws + OFF_WO);
  unsigned short* wF1  = (unsigned short*)(ws + OFF_FC1);
  unsigned short* wF2  = (unsigned short*)(ws + OFF_FC2);
  float* hsum = (float*)(ws + OFF_HS);
  float* lout = (float*)(ws + OFF_LO);

  // weight prep (fp32 -> bf16, zero-padded)
  auto grid = [](int n) { return (n + 255) / 256; };
  prep_bf16<<<grid(192 * 256), 256, 0, stream>>>(Wih,  wWih, 192, 256, 192, 256);
  prep_bf16<<<grid(192 * 64),  256, 0, stream>>>(Whh,  wWhh, 192, 64,  192, 64);
  prep_bf16<<<grid(PPAD * 64), 256, 0, stream>>>(Wp,   wWp,  Pn,  64,  PPAD, 64);
  prep_bf16<<<grid(64 * 192),  256, 0, stream>>>(Wo,   wWo,  64,  192, 64,  192);
  prep_bf16<<<grid(208 * 320), 256, 0, stream>>>(fc1w, wF1,  200, 320, 208, 320);
  prep_bf16<<<grid(80 * 224),  256, 0, stream>>>(fc2w, wF2,  80,  200, 80,  224);

  hist_sum_kernel<<<Bn, 128, 0, stream>>>(hitem, hcate, mask, emb, hsum);

  const size_t rnnLds = 306432;   // ~299 KB, fits 320 KB/WGP
  mimn_rnn_kernel<<<Bn / 16, 128, rnnLds, stream>>>(
      hitem, hcate, mask, emb, wWih, wWo, bih, bhh, bp, bo, M0, lout);

  const size_t mlpLds = 191488;
  mimn_mlp_kernel<<<Bn / 16, 128, mlpLds, stream>>>(
      item, cate, emb, hsum, lout, wF1, fc1b, p1a, fc2b, p2a, fc3w, fc3b,
      (float*)d_out);
}
